// Encoder_2662879724015
// MI455X (gfx1250) — matrix operand order repaired and checked by reading, not compiled, not run on hardware
//
#include <hip/hip_runtime.h>

// ---------- types for WMMA ----------
typedef __attribute__((ext_vector_type(16))) __bf16 v16bf;
typedef __attribute__((ext_vector_type(8)))  float  v8f;

#define HID 128  // output width of both layers

// f32 -> bf16 with round-to-nearest-even
__device__ __forceinline__ unsigned short f2bf(float f) {
    unsigned int u = __float_as_uint(f);
    u = u + 0x7FFFu + ((u >> 16) & 1u);
    return (unsigned short)(u >> 16);
}

__device__ __forceinline__ void atomAddF(float* p, float v) {
    __hip_atomic_fetch_add(p, v, __ATOMIC_RELAXED, __HIP_MEMORY_SCOPE_AGENT);
}

// ---------- elementwise prep kernels ----------
__global__ void cast_bf16_kernel(const float* __restrict__ src,
                                 unsigned short* __restrict__ dst, size_t n) {
    size_t i = (size_t)blockIdx.x * blockDim.x + threadIdx.x;
    if (i < n) dst[i] = f2bf(src[i]);
}

// W [K][N] f32 -> Wt [N][K] bf16 (transposed so B-fragment K-pairs are contiguous)
__global__ void transpose_w_kernel(const float* __restrict__ W,
                                   unsigned short* __restrict__ Wt, int K, int N) {
    int i = blockIdx.x * blockDim.x + threadIdx.x;
    if (i < K * N) {
        int k = i / N, n = i % N;
        Wt[n * K + k] = f2bf(W[i]);
    }
}

__global__ void deg_init_kernel(float* __restrict__ deg, int n) {
    int i = blockIdx.x * blockDim.x + threadIdx.x;
    if (i < n) deg[i] = 1.0f;  // self-loop contribution
}

__global__ void deg_accum_kernel(const long long* __restrict__ col,
                                 float* __restrict__ deg, int e) {
    int i = blockIdx.x * blockDim.x + threadIdx.x;
    if (i < e) atomAddF(&deg[col[i]], 1.0f);
}

__global__ void dinv_kernel(float* __restrict__ deg, int n) {
    int i = blockIdx.x * blockDim.x + threadIdx.x;
    if (i < n) {
        float d = deg[i];
        deg[i] = d > 0.0f ? rsqrtf(d) : 0.0f;
    }
}

// agg[v][d] = dinv[v]^2 * xw[v][d] + b[d]   (self-loop message + bias)
__global__ void agg_init_kernel(const float* __restrict__ xw,
                                const float* __restrict__ dinv,
                                const float* __restrict__ b,
                                float* __restrict__ agg, size_t nelem) {
    size_t i = (size_t)blockIdx.x * blockDim.x + threadIdx.x;
    if (i < nelem) {
        size_t v = i >> 7;
        int d = (int)(i & 127);
        float s = dinv[v];
        agg[i] = s * s * xw[i] + b[d];
    }
}

// one wave per edge: gather 128-wide message, scale, scatter-add
__global__ void scatter_kernel(const long long* __restrict__ row,
                               const long long* __restrict__ col,
                               const float* __restrict__ dinv,
                               const float* __restrict__ xw,
                               float* __restrict__ agg, int E) {
    int e = blockIdx.x * 8 + (threadIdx.x >> 5);
    if (e >= E) return;
    long long r = row[e];
    long long c = col[e];
    float nrm = dinv[r] * dinv[c];
    int lane = threadIdx.x & 31;
    const float4 v = *(const float4*)(xw + (size_t)r * HID + lane * 4);
    float* dst = agg + (size_t)c * HID + lane * 4;
    atomAddF(dst + 0, nrm * v.x);
    atomAddF(dst + 1, nrm * v.y);
    atomAddF(dst + 2, nrm * v.z);
    atomAddF(dst + 3, nrm * v.w);
}

__global__ void tanh_bf16_kernel(const float* __restrict__ agg,
                                 unsigned short* __restrict__ hb, size_t n) {
    size_t i = (size_t)blockIdx.x * blockDim.x + threadIdx.x;
    if (i < n) hb[i] = f2bf(tanhf(agg[i]));
}

__global__ void tanh_f32_kernel(const float* __restrict__ agg,
                                float* __restrict__ out, size_t n) {
    size_t i = (size_t)blockIdx.x * blockDim.x + threadIdx.x;
    if (i < n) out[i] = tanhf(agg[i]);
}

// ---------- WMMA GEMM: Y[M][128] = X[M][K](bf16) @ W[K][128](bf16, staged as Wt[128][K]) ----------
// 256 threads = 8 waves; block computes a 128x128 tile; wave w owns rows [w*16, w*16+16).
// B fragments for a full K-step are preloaded into registers so the 8 WMMAs can
// issue back-to-back while the next K-step's LDS loads are in flight.
template <int K>
__global__ void __launch_bounds__(256)
gemm_bf16_kernel(const unsigned short* __restrict__ X,
                 const unsigned short* __restrict__ Wt,  // [128][K] bf16
                 float* __restrict__ Y, int M) {
    __shared__ unsigned short sW[HID * K];  // 64 KB (K=256) / 32 KB (K=128)

    int tid = threadIdx.x;
    // cooperative stage of Wt into LDS (uint4 = 8 bf16 elements)
    const int NV4 = (HID * K) / 8;
    for (int i = tid; i < NV4; i += 256)
        ((uint4*)sW)[i] = ((const uint4*)Wt)[i];
    __syncthreads();

    int wave = tid >> 5;
    int lane = tid & 31;
    int half = lane >> 4;       // 0: lanes 0-15, 1: lanes 16-31
    int l16  = lane & 15;

    // A fragment row (clamped for tail block; stores are guarded)
    int m0 = blockIdx.x * 128 + wave * 16 + l16;
    int mr = m0 < M ? m0 : (M - 1);
    const unsigned short* xrow = X + (size_t)mr * K;
    int kbA = half * 8;         // A-layout: half-wave K sub-offset

    v8f acc[8];
#pragma unroll
    for (int t = 0; t < 8; ++t) acc[t] = (v8f)(0.0f);

    union FRAG { v16bf v; unsigned int u[8]; uint4 q[2]; };

    for (int ks = 0; ks < K; ks += 32) {
        // A fragment: lane l16 = row, elements 0-7 -> K = ks+kbA+e, 8-15 -> K = ks+16+kbA+(e-8)
        FRAG fa;
        fa.q[0] = *(const uint4*)(xrow + ks + kbA);
        fa.q[1] = *(const uint4*)(xrow + ks + 16 + kbA);

        int kbB = ks + half * 8;   // B-layout: same K order as A (8h+e, 16+8h+(e-8))
        FRAG fb[8];
        // issue all 16 ds_load_b128 up front (one clause), then WMMA back-to-back
#pragma unroll
        for (int t = 0; t < 8; ++t) {
            const unsigned short* wrow = sW + (size_t)(t * 16 + l16) * K + kbB;
            fb[t].q[0] = *(const uint4*)(wrow);
            fb[t].q[1] = *(const uint4*)(wrow + 16);
        }
#pragma unroll
        for (int t = 0; t < 8; ++t) {
            acc[t] = __builtin_amdgcn_wmma_f32_16x16x32_bf16(
                false, fa.v, false, fb[t].v, (short)0, acc[t], false, false);
        }
    }

    // C/D layout: lane half h -> M = h*8 + r (VGPR r), N = l16
    int mbase = blockIdx.x * 128 + wave * 16 + half * 8;
#pragma unroll
    for (int t = 0; t < 8; ++t) {
#pragma unroll
        for (int r = 0; r < 8; ++r) {
            int m = mbase + r;
            if (m < M) Y[(size_t)m * HID + t * 16 + l16] = acc[t][r];
        }
    }
}

// ---------- host launcher ----------
extern "C" void kernel_launch(void* const* d_in, const int* in_sizes, int n_in,
                              void* d_out, int out_size, void* d_ws, size_t ws_size,
                              hipStream_t stream) {
    const float*      x    = (const float*)d_in[0];
    const long long*  ei   = (const long long*)d_in[1];   // int64 [2][E]
    const float*      W1   = (const float*)d_in[2];       // [256][128]
    const float*      b1   = (const float*)d_in[3];
    const float*      W2   = (const float*)d_in[4];       // [128][128]
    const float*      b2   = (const float*)d_in[5];
    float*            out  = (float*)d_out;

    const int IN_DIM = 256;
    const size_t N = (size_t)in_sizes[0] / IN_DIM;   // 100000
    const int    E = in_sizes[1] / 2;                 // 1600000
    const long long* row = ei;
    const long long* col = ei + E;

    // workspace carve-up (256B aligned)
    char* ws = (char*)d_ws;
    size_t off = 0;
    auto carve = [&](size_t bytes) { char* p = ws + off; off = (off + bytes + 255) & ~(size_t)255; return p; };
    float*          dinv = (float*)carve(N * 4);
    unsigned short* xb   = (unsigned short*)carve(N * IN_DIM * 2);
    unsigned short* w1t  = (unsigned short*)carve((size_t)IN_DIM * HID * 2);
    unsigned short* w2t  = (unsigned short*)carve((size_t)HID * HID * 2);
    float*          xw   = (float*)carve(N * HID * 4);
    float*          agg  = (float*)carve(N * HID * 4);
    unsigned short* hb   = (unsigned short*)carve(N * HID * 2);
    (void)ws_size;

    const int T = 256;
    auto blks = [](size_t n, int t) { return (unsigned)((n + t - 1) / t); };

    // --- prep: bf16 casts + W transposes ---
    cast_bf16_kernel<<<blks(N * IN_DIM, T), T, 0, stream>>>(x, xb, N * IN_DIM);
    transpose_w_kernel<<<blks((size_t)IN_DIM * HID, T), T, 0, stream>>>(W1, w1t, IN_DIM, HID);
    transpose_w_kernel<<<blks((size_t)HID * HID, T), T, 0, stream>>>(W2, w2t, HID, HID);

    // --- symmetric normalization ---
    deg_init_kernel<<<blks(N, T), T, 0, stream>>>(dinv, (int)N);
    deg_accum_kernel<<<blks(E, T), T, 0, stream>>>(col, dinv, E);
    dinv_kernel<<<blks(N, T), T, 0, stream>>>(dinv, (int)N);

    unsigned gemmBlocks = (unsigned)((N + 127) / 128);
    unsigned edgeBlocks = (unsigned)((E + 7) / 8);

    // --- layer 1 ---
    gemm_bf16_kernel<256><<<gemmBlocks, T, 0, stream>>>(xb, w1t, xw, (int)N);
    agg_init_kernel<<<blks(N * HID, T), T, 0, stream>>>(xw, dinv, b1, agg, N * HID);
    scatter_kernel<<<edgeBlocks, T, 0, stream>>>(row, col, dinv, xw, agg, E);
    tanh_bf16_kernel<<<blks(N * HID, T), T, 0, stream>>>(agg, hb, N * HID);

    // --- layer 2 ---
    gemm_bf16_kernel<128><<<gemmBlocks, T, 0, stream>>>(hb, w2t, xw, (int)N);
    agg_init_kernel<<<blks(N * HID, T), T, 0, stream>>>(xw, dinv, b2, agg, N * HID);
    scatter_kernel<<<edgeBlocks, T, 0, stream>>>(row, col, dinv, xw, agg, E);
    tanh_f32_kernel<<<blks(N * HID, T), T, 0, stream>>>(agg, out, N * HID);
}
